// GAT_73658689126420
// MI455X (gfx1250) — matrix operand order repaired and checked by reading, not compiled, not run on hardware
//
#include <hip/hip_runtime.h>
#include <hip/hip_bf16.h>

typedef __attribute__((ext_vector_type(16))) _Float16 v16h;
typedef __attribute__((ext_vector_type(8)))  float    v8f;

// ---------------------------------------------------------------------------
// A-fragment loader, per CDNA5 ISA 7.12.2 (wave32, f16 16x16x32).
// Lane l<16, row=l: elems 0-7 = K 0..7, elems 8-15 = K 16..23 (of this kk slab);
// lane l>=16 same row: K 8..15 and K 24..31. All as two pairs of float4 loads.
// ---------------------------------------------------------------------------
__device__ __forceinline__ v16h load_a_frag(const float* __restrict__ A,
                                            int rowBase, int lane, int kk, int lda) {
    const int row  = rowBase + (lane & 15);
    const int koff = (lane >> 4) ? 8 : 0;
    const float* base = A + (size_t)row * lda + kk + koff;
    const float4 x0 = ((const float4*)base)[0];
    const float4 x1 = ((const float4*)base)[1];
    const float4 y0 = ((const float4*)(base + 16))[0];
    const float4 y1 = ((const float4*)(base + 16))[1];
    v16h a;
    a[0]  = (_Float16)x0.x; a[1]  = (_Float16)x0.y; a[2]  = (_Float16)x0.z; a[3]  = (_Float16)x0.w;
    a[4]  = (_Float16)x1.x; a[5]  = (_Float16)x1.y; a[6]  = (_Float16)x1.z; a[7]  = (_Float16)x1.w;
    a[8]  = (_Float16)y0.x; a[9]  = (_Float16)y0.y; a[10] = (_Float16)y0.z; a[11] = (_Float16)y0.w;
    a[12] = (_Float16)y1.x; a[13] = (_Float16)y1.y; a[14] = (_Float16)y1.z; a[15] = (_Float16)y1.w;
    return a;
}

// ---------------------------------------------------------------------------
// C = A[nrows x 64] * W[64 x NCOLS] (+ bias).
// W is pre-swizzled into WMMA B-fragment layout in LDS once per block; each
// wave hoists its fragments to registers and then loops over 16-row tiles.
// nrows must be a multiple of 16 (N = 100000 = 6250 * 16).
// ---------------------------------------------------------------------------
template <int NCOLS>
__global__ void gemm_wmma_kernel(const float* __restrict__ A,
                                 const float* __restrict__ W,
                                 const float* __restrict__ bias,
                                 float* __restrict__ C, int nrows) {
    constexpr int NT    = NCOLS / 16;   // column tiles
    constexpr int NFRAG = 2 * NT;       // (kk in {0,32}) x NT

    __shared__ v16h wlds[NFRAG * 32];   // frag-major, per-lane 16 f16 (32B aligned)
    _Float16* wf = (_Float16*)wlds;

    const int tid  = threadIdx.x;
    const int lane = tid & 31;
    const int wave = tid >> 5;

    // Cooperative, coalesced fill + swizzle of W (64 x NCOLS f32 -> f16 frags).
    // Element (k, col) -> frag = (k>>5)*NT + (col>>4),
//                     lane = (col&15) | ((k>>3)&1)<<4,  elem = (k&7) | ((k>>4)&1)<<3.
    for (int idx = tid; idx < 64 * NCOLS; idx += 256) {
        const int k    = idx / NCOLS;
        const int col  = idx - k * NCOLS;
        const int frag = (k >> 5) * NT + (col >> 4);
        const int l    = (col & 15) | (((k >> 3) & 1) << 4);
        const int e    = (k & 7) | (((k >> 4) & 1) << 3);
        wf[(frag * 32 + l) * 16 + e] = (_Float16)W[idx];
    }
    __syncthreads();

    // B fragments: register-resident for the whole tile loop.
    v16h bfrag[NFRAG];
#pragma unroll
    for (int f = 0; f < NFRAG; ++f) bfrag[f] = wlds[f * 32 + lane];

    const int col0    = lane & 15;
    const int rowOff  = (lane >> 4) ? 8 : 0;
    const int nTiles  = nrows >> 4;
    const int stride  = gridDim.x * (blockDim.x >> 5);

    for (int tile = blockIdx.x * (blockDim.x >> 5) + wave; tile < nTiles; tile += stride) {
        const int rowBase = tile * 16;
        v8f acc[NT] = {};
#pragma unroll
        for (int kk = 0; kk < 2; ++kk) {
            const v16h a = load_a_frag(A, rowBase, lane, kk * 32, 64);
#pragma unroll
            for (int t = 0; t < NT; ++t)
                acc[t] = __builtin_amdgcn_wmma_f32_16x16x32_f16(
                    false, a, false, bfrag[kk * NT + t], (short)0, acc[t], false, false);
        }
        // D layout: VGPR v -> row rowBase + rowOff + v, col = (lane&15) + 16t.
#pragma unroll
        for (int t = 0; t < NT; ++t) {
            const int col = t * 16 + col0;
            const float bv = bias ? bias[col] : 0.0f;
#pragma unroll
            for (int v = 0; v < 8; ++v)
                C[(size_t)(rowBase + rowOff + v) * NCOLS + col] = acc[t][v] + bv;
        }
    }
}

// ---------------------------------------------------------------------------
// Attention pipeline kernels (f32; gathers/atomics resolve in the 192MB L2)
// ---------------------------------------------------------------------------
__global__ void alpha_kernel(const float* __restrict__ h,
                             const float* __restrict__ a_s,
                             const float* __restrict__ a_d,
                             float* __restrict__ as_out, float* __restrict__ ad_out,
                             int n) {
    const int i = blockIdx.x * blockDim.x + threadIdx.x;
    if (i >= n) return;
    const float* hp = h + (size_t)i * 64;
    float s = 0.0f, d = 0.0f;
#pragma unroll 8
    for (int f = 0; f < 64; ++f) { const float hv = hp[f]; s += hv * a_s[f]; d += hv * a_d[f]; }
    as_out[i] = s;
    ad_out[i] = d;
}

__global__ void init_kernel(float* __restrict__ agg, float* __restrict__ m,
                            float* __restrict__ denom, int n) {
    const int i = blockIdx.x * blockDim.x + threadIdx.x;
    if (i < n) { m[i] = __uint_as_float(0xff800000u); denom[i] = 0.0f; }
    if (i < n * 64) agg[i] = 0.0f;
}

__device__ __forceinline__ void atomic_max_f32(float* addr, float v) {
    if (v >= 0.0f) atomicMax((int*)addr, __float_as_int(v));
    else           atomicMin((unsigned int*)addr, __float_as_uint(v));
}

__device__ __forceinline__ void edge_ends(const int* __restrict__ src,
                                          const int* __restrict__ dst,
                                          int e, int E, int& s, int& d) {
    if (e < E) { s = src[e]; d = dst[e]; }
    else       { s = e - E;  d = e - E;  }
}

__device__ __forceinline__ float leaky02(float v) { return v > 0.0f ? v : 0.2f * v; }

__global__ void edge_max_kernel(const int* __restrict__ src, const int* __restrict__ dst,
                                const float* __restrict__ as_, const float* __restrict__ ad_,
                                float* __restrict__ m, int E, int EP) {
    const int e = blockIdx.x * blockDim.x + threadIdx.x;
    if (e >= EP) return;
    int s, d; edge_ends(src, dst, e, E, s, d);
    atomic_max_f32(m + d, leaky02(as_[s] + ad_[d]));
}

__global__ void edge_sum_kernel(const int* __restrict__ src, const int* __restrict__ dst,
                                const float* __restrict__ as_, const float* __restrict__ ad_,
                                const float* __restrict__ m, float* __restrict__ denom,
                                int E, int EP) {
    const int e = blockIdx.x * blockDim.x + threadIdx.x;
    if (e >= EP) return;
    int s, d; edge_ends(src, dst, e, E, s, d);
    atomicAdd(denom + d, __expf(leaky02(as_[s] + ad_[d]) - m[d]));
}

// One wave per edge: lane handles features lane and lane+32 (coalesced 128B rows);
// per-edge scalar reads broadcast across the wave.
__global__ void edge_scatter_kernel(const int* __restrict__ src, const int* __restrict__ dst,
                                    const float* __restrict__ as_, const float* __restrict__ ad_,
                                    const float* __restrict__ m, const float* __restrict__ denom,
                                    const float* __restrict__ h, float* __restrict__ out,
                                    int E, int EP) {
    const int lane = threadIdx.x & 31;
    const int wave = threadIdx.x >> 5;
    const int e = blockIdx.x * (blockDim.x >> 5) + wave;
    if (e >= EP) return;
    int s, d; edge_ends(src, dst, e, E, s, d);
    const float coef = __expf(leaky02(as_[s] + ad_[d]) - m[d]) / (denom[d] + 1e-16f);
    const float h0 = h[(size_t)s * 64 + lane];
    const float h1 = h[(size_t)s * 64 + 32 + lane];
    atomicAdd(out + (size_t)d * 64 + lane,      coef * h0);
    atomicAdd(out + (size_t)d * 64 + 32 + lane, coef * h1);
}

__global__ void bias_relu_kernel(float* __restrict__ h, const float* __restrict__ b, int total) {
    const int i = blockIdx.x * blockDim.x + threadIdx.x;
    if (i >= total) return;
    const float v = h[i] + b[i & 63];
    h[i] = v > 0.0f ? v : 0.0f;
}

// ---------------------------------------------------------------------------
// Host launcher
// ---------------------------------------------------------------------------
extern "C" void kernel_launch(void* const* d_in, const int* in_sizes, int n_in,
                              void* d_out, int out_size, void* d_ws, size_t ws_size,
                              hipStream_t stream) {
    const float* x   = (const float*)d_in[0];
    const int*   ei  = (const int*)d_in[1];
    const int N  = in_sizes[0] / 64;
    const int E  = in_sizes[1] / 2;
    const int EP = E + N;                 // with self-loops
    const int* srcA = ei;
    const int* dstA = ei + E;

    // Workspace layout (floats): H (N*64), G (N*64), as, ad, m, denom (N each)
    float* H    = (float*)d_ws;
    float* G    = H + (size_t)N * 64;
    float* as_  = G + (size_t)N * 64;
    float* ad_  = as_ + N;
    float* m_   = ad_ + N;
    float* den_ = m_ + N;

    const int TPB = 256;
    const int rowTiles   = (N + 15) / 16;
    int gemmBlocks = (rowTiles + 7) / 8;                // 8 waves / block
    if (gemmBlocks > 256) gemmBlocks = 256;             // persistent waves, ~3 tiles each
    const int nodeBlocks = (N + TPB - 1) / TPB;
    const int featBlocks = (N * 64 + TPB - 1) / TPB;
    const int edgeBlocks = (EP + TPB - 1) / TPB;
    const int scatBlocks = (EP + 7) / 8;                // wave per edge

    const float* P = x;
    for (int L = 0; L < 3; ++L) {
        const float* W  = (const float*)d_in[2 + L * 4];
        const float* aS = (const float*)d_in[3 + L * 4];
        const float* aD = (const float*)d_in[4 + L * 4];
        const float* b  = (const float*)d_in[5 + L * 4];

        gemm_wmma_kernel<64><<<gemmBlocks, TPB, 0, stream>>>(P, W, nullptr, H, N);
        alpha_kernel<<<nodeBlocks, TPB, 0, stream>>>(H, aS, aD, as_, ad_, N);
        init_kernel<<<featBlocks, TPB, 0, stream>>>(G, m_, den_, N);
        edge_max_kernel<<<edgeBlocks, TPB, 0, stream>>>(srcA, dstA, as_, ad_, m_, E, EP);
        edge_sum_kernel<<<edgeBlocks, TPB, 0, stream>>>(srcA, dstA, as_, ad_, m_, den_, E, EP);
        edge_scatter_kernel<<<scatBlocks, TPB, 0, stream>>>(srcA, dstA, as_, ad_, m_, den_, H, G, E, EP);
        bias_relu_kernel<<<featBlocks, TPB, 0, stream>>>(G, b, N * 64);
        P = G;   // aggregated output feeds next layer (ping-pong with H)
    }

    const float* Wl = (const float*)d_in[14];
    const float* bl = (const float*)d_in[15];
    gemm_wmma_kernel<32><<<gemmBlocks, TPB, 0, stream>>>(P, Wl, bl, (float*)d_out, N);
}
